// BGRU_CRF_54322746360280
// MI455X (gfx1250) — matrix operand order repaired and checked by reading, not compiled, not run on hardware
//
#include <hip/hip_runtime.h>
#include <cstdint>
#include <cstddef>

#define SEQL   4096
#define EMBD   256
#define HIDD   512
#define H3     1536
#define NTAGS  16
#define START_TAG 14
#define STOP_TAG  15

typedef __attribute__((ext_vector_type(16))) __bf16        v16bf;
typedef __attribute__((ext_vector_type(8)))  float         v8f;
typedef __attribute__((ext_vector_type(2)))  float         v2f;
typedef __attribute__((ext_vector_type(4)))  unsigned int  v4u;
typedef __attribute__((ext_vector_type(8)))  int           v8i;
typedef __attribute__((ext_vector_type(4)))  int           v4i;

static __device__ __forceinline__ unsigned short f2bf(float f) {
  union { float f; unsigned u; } c; c.f = f;
  unsigned u = c.u;
  u += 0x7FFFu + ((u >> 16) & 1u);      // round-to-nearest-even
  return (unsigned short)(u >> 16);
}
static __device__ __forceinline__ float bf_lo(unsigned u) {
  return __uint_as_float(u << 16);
}
static __device__ __forceinline__ float bf_hi(unsigned u) {
  return __uint_as_float(u & 0xFFFF0000u);
}

// ---------------------------------------------------------------------------
// TDM: DMA a 2D bf16 tile (rows x cols, row stride in elements) from global
// memory into LDS.  D# layout per CDNA5 ISA ch.8:
//   group0: [1:0]=count=1, [63:32]=lds_addr, [120:64]=global_addr, [127:126]=2
//   group1: [17:16]=data_size(1->2B), [79:48]=tensor_dim0, [111:80]=tensor_dim1,
//           [127:112]=tile_dim0, [143:128]=tile_dim1, [207:160]=dim0_stride
// This toolchain exposes the 6-arg builtin:
//   (uint32x4 g0, int32x8 g1, int32x4, int32x4, int32x8, i32 cpol)
// ---------------------------------------------------------------------------
#if __has_builtin(__builtin_amdgcn_tensor_load_to_lds)
#define HAVE_TDM 1
static __device__ __forceinline__ void tdm_load_2d_bf16(
    unsigned lds_addr, const void* gptr, unsigned rows, unsigned cols,
    unsigned row_stride) {
  const unsigned long long ga = (unsigned long long)gptr;
  v4u g0;
  g0[0] = 1u;                                         // count=1, user mode
  g0[1] = lds_addr;                                   // LDS byte address
  g0[2] = (unsigned)(ga & 0xFFFFFFFFu);               // global_addr[31:0]
  g0[3] = (unsigned)((ga >> 32) & 0x01FFFFFFu)        // global_addr[56:32]
          | (2u << 30);                               // type = 2 ("image")
  v8i g1;
  g1[0] = (int)(1u << 16);                            // data_size = 1 -> 2 bytes
  g1[1] = (int)(cols << 16);                          // tensor_dim0[15:0]
  g1[2] = (int)(rows << 16);                          // dim0 hi=0 | tensor_dim1[15:0]
  g1[3] = (int)(cols << 16);                          // dim1 hi=0 | tile_dim0
  g1[4] = (int)rows;                                  // tile_dim1 | tile_dim2=0
  g1[5] = (int)row_stride;                            // tensor_dim0_stride[31:0]
  g1[6] = 0;
  g1[7] = 0;
  const v4i z4 = {};
  const v8i z8 = {};
  __builtin_amdgcn_tensor_load_to_lds(g0, g1, z4, z4, z8, 0);
}
#else
#define HAVE_TDM 0
#endif

// ---------------------------------------------------------------------------
// 1) Embedding gather -> bf16   (xe[t][e] = bf16(emb[sentence[t]][e]))
// ---------------------------------------------------------------------------
__global__ void k_gather(const int* __restrict__ sent,
                         const float* __restrict__ emb,
                         unsigned short* __restrict__ xe) {
  const int t = blockIdx.x;
  const int e = threadIdx.x;
  xe[(size_t)t * EMBD + e] = f2bf(emb[(size_t)sent[t] * EMBD + e]);
}

// ---------------------------------------------------------------------------
// 2) Generic f32 -> bf16 weight conversion
// ---------------------------------------------------------------------------
__global__ void k_cvt(const float* __restrict__ src,
                      unsigned short* __restrict__ dst, int n) {
  const int i = blockIdx.x * blockDim.x + threadIdx.x;
  if (i < n) dst[i] = f2bf(src[i]);
}

// ---------------------------------------------------------------------------
// 3) Input projection: xp[dir][t][1536] = x @ Wih_dir.T + bih_dir
//    TDM stages A (16x256, shared) and per-wave B (16x256) strips into LDS;
//    K-loop of v_wmma_f32_16x16x32_bf16 reads fragments via ds_load_b128.
//    A 16x32 frag: lane(m=lane&15,hi); halves 0..7 => K=hi*8+h, 8..15 => K=16+hi*8+h-8.
//    B 32x16 frag: lane(n=lane&15,hi); halves 0..7 => K=hi*8+h, 8..15 => K=16+hi*8+h-8.
// ---------------------------------------------------------------------------
__global__ void __launch_bounds__(128)
k_xproj(const unsigned short* __restrict__ xe,
        const unsigned short* __restrict__ wih,
        const float* __restrict__ bih_f,
        const float* __restrict__ bih_b,
        float* __restrict__ xp) {
  const int lane   = threadIdx.x & 31;
  const int wave   = threadIdx.x >> 5;       // 0..3
  const int nc     = blockIdx.x * 4 + wave;  // col tile 0..95
  const int tr     = blockIdx.y;             // row tile 0..255
  const int dir    = blockIdx.z;             // 0 fwd, 1 bwd
  const int laneHi = lane >> 4;
  const int lm     = lane & 15;

  __shared__ __align__(16) unsigned short smA[16 * EMBD];      // 8 KB
  __shared__ __align__(16) unsigned short smB[4][16 * EMBD];   // 32 KB

  const unsigned short* W   = wih + (size_t)dir * H3 * EMBD;
  const float*          bih = dir ? bih_b : bih_f;
  const unsigned short* ga  = xe + (size_t)tr * 16 * EMBD;
  const unsigned short* gb  = W  + (size_t)nc * 16 * EMBD;

#if HAVE_TDM
  if (wave == 0)
    tdm_load_2d_bf16((unsigned)(uintptr_t)(void*)smA, ga, 16, EMBD, EMBD);
  tdm_load_2d_bf16((unsigned)(uintptr_t)(void*)smB[wave], gb, 16, EMBD, EMBD);
  __builtin_amdgcn_s_wait_tensorcnt(0);      // own TDM ops complete
  __syncthreads();                           // wave0's A strip visible to all
#else
  for (int i = threadIdx.x; i < 16 * EMBD / 8; i += blockDim.x)
    ((uint4*)smA)[i] = ((const uint4*)ga)[i];
  for (int i = lane; i < 16 * EMBD / 8; i += 32)
    ((uint4*)smB[wave])[i] = ((const uint4*)gb)[i];
  __syncthreads();
#endif

  const unsigned short* arow = &smA[lm * EMBD];
  const unsigned short* brow = &smB[wave][lm * EMBD];

  union { v16bf bf; uint4 q[2]; } A, B;
  v8f acc = {};
#pragma unroll
  for (int ks = 0; ks < EMBD / 32; ++ks) {
    const int k0 = ks * 32;
    A.q[0] = *(const uint4*)(arow + k0 + laneHi * 8);
    A.q[1] = *(const uint4*)(arow + k0 + 16 + laneHi * 8);
    B.q[0] = *(const uint4*)(brow + k0 + laneHi * 8);
    B.q[1] = *(const uint4*)(brow + k0 + 16 + laneHi * 8);
    acc = __builtin_amdgcn_wmma_f32_16x16x32_bf16(
        false, A.bf, false, B.bf, (short)0, acc, false, false);
  }

  float* out = xp + ((size_t)dir * SEQL + (size_t)tr * 16) * H3 + nc * 16;
  const float b = bih[nc * 16 + lm];
#pragma unroll
  for (int r = 0; r < 8; ++r) {
    const int m = r + 8 * laneHi;            // C/D layout: VGPR r, laneHi selects M/M+8
    out[(size_t)m * H3 + lm] = acc[r] + b;
  }
}

// ---------------------------------------------------------------------------
// 4) Recurrent scan, one workgroup (512 threads) per direction.
//    h kept in LDS; Whh streamed as packed bf16 from L2 each step.
// ---------------------------------------------------------------------------
__global__ void k_scan(const float* __restrict__ xp,
                       const unsigned short* __restrict__ whh,
                       const float* __restrict__ bhh_f,
                       const float* __restrict__ bhh_b,
                       float* __restrict__ hs) {
  const int dir = blockIdx.x;
  const int u   = threadIdx.x;               // hidden unit 0..511
  __shared__ __align__(16) float h[HIDD];
  h[u] = 0.0f;
  __syncthreads();

  const unsigned short* W   = whh + (size_t)dir * H3 * HIDD;
  const float*          bhh = dir ? bhh_b : bhh_f;
  const uint2* wr = (const uint2*)(W + (size_t)u * HIDD);
  const uint2* wz = (const uint2*)(W + (size_t)(HIDD + u) * HIDD);
  const uint2* wn = (const uint2*)(W + (size_t)(2 * HIDD + u) * HIDD);
  const float4* h4 = (const float4*)h;
  const float brr = bhh[u], bzz = bhh[HIDD + u], bnn = bhh[2 * HIDD + u];

  for (int t = 0; t < SEQL; ++t) {
    const int ti = dir ? (SEQL - 1 - t) : t;
    const float* xrow = xp + ((size_t)dir * SEQL + ti) * H3;

    float hr = brr, hz = bzz, hn = bnn;
#pragma unroll 4
    for (int k = 0; k < HIDD / 4; ++k) {
      const float4 hv = h4[k];
      const uint2 a = wr[k], b = wz[k], c = wn[k];
      hr += bf_lo(a.x) * hv.x + bf_hi(a.x) * hv.y +
            bf_lo(a.y) * hv.z + bf_hi(a.y) * hv.w;
      hz += bf_lo(b.x) * hv.x + bf_hi(b.x) * hv.y +
            bf_lo(b.y) * hv.z + bf_hi(b.y) * hv.w;
      hn += bf_lo(c.x) * hv.x + bf_hi(c.x) * hv.y +
            bf_lo(c.y) * hv.z + bf_hi(c.y) * hv.w;
    }

    const float xr = xrow[u], xz = xrow[HIDD + u], xn = xrow[2 * HIDD + u];
    const float r = 1.0f / (1.0f + __expf(-(xr + hr)));
    const float z = 1.0f / (1.0f + __expf(-(xz + hz)));
    const float n = tanhf(xn + r * hn);
    const float hold = h[u];
    const float hnew = (1.0f - z) * n + z * hold;

    __syncthreads();
    h[u] = hnew;
    hs[((size_t)dir * SEQL + ti) * HIDD + u] = hnew;
    __syncthreads();

    if (t + 1 < SEQL) {                      // global_prefetch next timestep xp row
      const int tn = dir ? (SEQL - 2 - t) : (t + 1);
      __builtin_prefetch(xp + ((size_t)dir * SEQL + tn) * H3 + u, 0, 0);
    }
  }
}

// ---------------------------------------------------------------------------
// 5) Output projection: feats = (h_f + h_b) @ Wout.T + bout
//    v_wmma_f32_16x16x4_f32 (pure f32 — preserves Viterbi fidelity).
// ---------------------------------------------------------------------------
__global__ void k_feats(const float* __restrict__ hs,
                        const float* __restrict__ wout,
                        const float* __restrict__ bout,
                        float* __restrict__ feats) {
  const int lane   = threadIdx.x & 31;
  const int laneHi = lane >> 4;
  const int lm     = lane & 15;
  const int tr     = blockIdx.x;             // row tile 0..255

  const float* hf = hs + (size_t)(tr * 16 + lm) * HIDD;
  const float* hb = hs + ((size_t)SEQL + tr * 16 + lm) * HIDD;
  const float* wr = wout + (size_t)lm * HIDD;

  v8f acc = {};
  for (int ks = 0; ks < HIDD / 4; ++ks) {
    const int k0 = ks * 4 + laneHi * 2;
    v2f a, b;
    a[0] = hf[k0] + hb[k0];
    a[1] = hf[k0 + 1] + hb[k0 + 1];
    b[0] = wr[k0];
    b[1] = wr[k0 + 1];
    acc = __builtin_amdgcn_wmma_f32_16x16x4_f32(
        false, a, false, b, (short)0, acc, false, false);
  }

  const float bo = bout[lm];
#pragma unroll
  for (int r = 0; r < 8; ++r)
    feats[(size_t)(tr * 16 + r + 8 * laneHi) * NTAGS + lm] = acc[r] + bo;
}

// ---------------------------------------------------------------------------
// 6) Viterbi DP + backtrack.  One wave32; lane = next-tag.
//    d_out[0] = score; d_out[1..4096] = path (as float).
// ---------------------------------------------------------------------------
__global__ void k_viterbi(const float* __restrict__ feats,
                          const float* __restrict__ trans,
                          unsigned char* __restrict__ bp,
                          float* __restrict__ out) {
  __shared__ float fv[NTAGS];
  __shared__ float trl[NTAGS * NTAGS];
  const int lane = threadIdx.x;
  for (int i = lane; i < NTAGS * NTAGS; i += 32) trl[i] = trans[i];
  if (lane < NTAGS) fv[lane] = (lane == START_TAG) ? 0.0f : -1000.0f;
  __syncthreads();

  for (int t = 0; t < SEQL; ++t) {
    float m = -3.4e38f; int arg = 0;
    if (lane < NTAGS) {
#pragma unroll
      for (int p = 0; p < NTAGS; ++p) {
        const float s = fv[p] + trl[lane * NTAGS + p];
        if (s > m) { m = s; arg = p; }      // strict > keeps first max (argmax)
      }
    }
    __syncthreads();
    if (lane < NTAGS) {
      fv[lane] = m + feats[(size_t)t * NTAGS + lane];
      bp[(size_t)t * NTAGS + lane] = (unsigned char)arg;
    }
    __syncthreads();
  }

  if (lane == 0) {
    float best = -3.4e38f; int bi = 0;
    for (int j = 0; j < NTAGS; ++j) {
      const float s = fv[j] + trl[STOP_TAG * NTAGS + j];
      if (s > best) { best = s; bi = j; }
    }
    out[0] = best;
    out[SEQL] = (float)bi;                  // path[SEQ-1] = best tag
    int tag = bi;
    for (int t = SEQL - 1; t >= 1; --t) {   // path[t-1] = prevs[t]
      tag = bp[(size_t)t * NTAGS + tag];
      out[t] = (float)tag;
    }
  }
}

// ---------------------------------------------------------------------------
// Launch
// ---------------------------------------------------------------------------
extern "C" void kernel_launch(void* const* d_in, const int* in_sizes, int n_in,
                              void* d_out, int out_size, void* d_ws, size_t ws_size,
                              hipStream_t stream) {
  const int*   sent  = (const int*)  d_in[0];
  const float* emb   = (const float*)d_in[1];
  const float* Wih_f = (const float*)d_in[2];
  const float* Whh_f = (const float*)d_in[3];
  const float* bih_f = (const float*)d_in[4];
  const float* bhh_f = (const float*)d_in[5];
  const float* Wih_b = (const float*)d_in[6];
  const float* Whh_b = (const float*)d_in[7];
  const float* bih_b = (const float*)d_in[8];
  const float* bhh_b = (const float*)d_in[9];
  const float* Wout  = (const float*)d_in[10];
  const float* bout  = (const float*)d_in[11];
  const float* trans = (const float*)d_in[12];

  char* ws = (char*)d_ws;
  unsigned short* xe    = (unsigned short*)(ws);                       // 2,097,152 B
  unsigned short* wih   = (unsigned short*)(ws + 2097152);             // 1,572,864 B
  unsigned short* whh   = (unsigned short*)(ws + 3670016);             // 3,145,728 B
  float*          xp    = (float*)         (ws + 6815744);             // 50,331,648 B
  float*          hs    = (float*)         (ws + 57147392);            // 16,777,216 B
  unsigned char*  bp    = (unsigned char*) (ws + 73924608);            // 65,536 B
  float*          feats = (float*)         (ws + 73990144);            // 262,144 B

  // 1) embed gather -> bf16
  k_gather<<<SEQL, EMBD, 0, stream>>>(sent, emb, xe);

  // 2) weight conversions -> bf16
  {
    const int nWih = H3 * EMBD;   // per direction
    const int nWhh = H3 * HIDD;
    k_cvt<<<(nWih + 255) / 256, 256, 0, stream>>>(Wih_f, wih, nWih);
    k_cvt<<<(nWih + 255) / 256, 256, 0, stream>>>(Wih_b, wih + (size_t)nWih, nWih);
    k_cvt<<<(nWhh + 255) / 256, 256, 0, stream>>>(Whh_f, whh, nWhh);
    k_cvt<<<(nWhh + 255) / 256, 256, 0, stream>>>(Whh_b, whh + (size_t)nWhh, nWhh);
  }

  // 3) WMMA input projection (TDM-staged): 24 x (4 waves) col tiles, 256 row tiles, 2 dirs
  k_xproj<<<dim3(24, 256, 2), 128, 0, stream>>>(xe, wih, bih_f, bih_b, xp);

  // 4) serial recurrent scan, one WGP per direction
  k_scan<<<2, 512, 0, stream>>>(xp, whh, bhh_f, bhh_b, hs);

  // 5) WMMA output projection
  k_feats<<<SEQL / 16, 32, 0, stream>>>(hs, Wout, bout, feats);

  // 6) Viterbi
  k_viterbi<<<1, 32, 0, stream>>>(feats, trans, bp, (float*)d_out);
}
